// PPF_14336600834607
// MI455X (gfx1250) — matrix operand order repaired and checked by reading, not compiled, not run on hardware
//
#include <hip/hip_runtime.h>
#include <hip/hip_bf16.h>

typedef _Float16 half_t;
typedef _Float16 v16h __attribute__((ext_vector_type(16)));
typedef _Float16 v8h  __attribute__((ext_vector_type(8)));
typedef float    v8f  __attribute__((ext_vector_type(8)));
typedef int      v4i  __attribute__((vector_size(16)));

#define NB   4          // batch
#define NPTS 4096       // points
#define KNN  64         // neighbors
#define RAD2 0.04f
#define EPSV 1e-5f
#define PP   (NPTS * KNN)       // columns per batch = 262144
#define BP   (NB * PP)          // total columns     = 1048576

// Detect CDNA5 async copy-to-LDS path (toolchain dependent); guarded fallback.
#if defined(__AMDGCN__) && __has_builtin(__builtin_amdgcn_global_load_async_to_lds_b128)
#define HAVE_ASYNC_LDS 1
typedef __attribute__((address_space(1))) v4i* gv4i_ptr;
typedef __attribute__((address_space(3))) v4i* lv4i_ptr;
#endif

// ---------------- workspace layout (bytes) ----------------
static const size_t OFF_IDX   = 0;                        // B*N*K int32 (4MB)
static const size_t OFF_STATS = 4194304;                  // 32KB stats
static const size_t OFF_W     = 4194304 + 32768;          // padded f16 weights
static const size_t OFF_A     = 8388608;                  // F0 (BPx32) then X1 (BPx128)
static const size_t OFF_B     = 8388608 + 268435456ull;   // X0 (BPx64) then X2 (BPx256)

// ---------------- zero ----------------
__global__ void zero_kernel(float* p, int n) {
    int t = blockIdx.x * 256 + threadIdx.x;
    if (t < n) p[t] = 0.f;
}

// ---------------- weight pad/convert ----------------
__global__ void wpad_kernel(const float* __restrict__ W0, const float* __restrict__ W1,
                            const float* __restrict__ W2,
                            half_t* __restrict__ W0p, half_t* __restrict__ W1p,
                            half_t* __restrict__ W2p) {
    int t = blockIdx.x * 256 + threadIdx.x;
    if (t < 64 * 32) {
        int r = t / 32, c = t % 32;
        W0p[t] = (half_t)(c < 10 ? W0[r * 10 + c] : 0.f);
    } else if (t < 64 * 32 + 128 * 64) {
        int u = t - 64 * 32;
        W1p[u] = (half_t)W1[u];
    } else if (t < 64 * 32 + 128 * 64 + 256 * 128) {
        int u = t - 64 * 32 - 128 * 64;
        W2p[u] = (half_t)W2[u];
    }
}

// ---------------- radius-KNN, one wave32 per point ----------------
__global__ __launch_bounds__(256) void knn_kernel(const float* __restrict__ coords,
                                                  int* __restrict__ idxbuf) {
    const int wave = threadIdx.x >> 5;
    const int lane = threadIdx.x & 31;
    const int pt   = blockIdx.x * 8 + wave;          // [0, NB*NPTS)
    const int b    = pt >> 12;
    const int i    = pt & (NPTS - 1);
    const float* cb = coords + (size_t)b * 3 * NPTS;
    const float cx = cb[i], cy = cb[NPTS + i], cz = cb[2 * NPTS + i];
    const int base = pt * KNN;

    int found = 0;
    int firstj = 0;
    bool have_first = false;
    for (int j0 = 0; j0 < NPTS; j0 += 32) {
        int j = j0 + lane;
        float dx = cb[j] - cx, dy = cb[NPTS + j] - cy, dz = cb[2 * NPTS + j] - cz;
        float d2 = dx * dx + dy * dy + dz * dz;
        bool pred = (d2 <= RAD2);
        unsigned m = (unsigned)__ballot(pred);          // wave32: low 32 bits
        if (!have_first && m) { firstj = j0 + __ffs(m) - 1; have_first = true; }
        int rank = __popc(m & ((1u << lane) - 1u));
        if (pred && (found + rank) < KNN) idxbuf[base + found + rank] = j;
        found += __popc(m);
        if (found >= KNN) break;
    }
    if (found < KNN) {
        for (int t = found + lane; t < KNN; t += 32) idxbuf[base + t] = firstj;
    }
}

// ---------------- PPF features -> f16, padded to 32 ch ----------------
__device__ __forceinline__ float angle3(float ax, float ay, float az,
                                        float bx, float by, float bz) {
    float cx = ay * bz - az * by;
    float cy = az * bx - ax * bz;
    float cz = ax * by - ay * bx;
    float cn = sqrtf(cx * cx + cy * cy + cz * cz);
    float d  = ax * bx + ay * by + az * bz;
    return atan2f(cn, d);
}

__global__ __launch_bounds__(256) void features_kernel(const float* __restrict__ coords,
                                                       const float* __restrict__ feats,
                                                       const int* __restrict__ idx,
                                                       half_t* __restrict__ F) {
    long long gid = (long long)blockIdx.x * 256 + threadIdx.x;   // over BP
    long long bn  = gid / KNN;
    int n = (int)(bn % NPTS);
    int b = (int)(bn / NPTS);
    const float* cb = coords + (size_t)b * 3 * NPTS;
    const float* fb = feats + (size_t)b * 3 * NPTS;
    int j = idx[gid];

    float cx = cb[n], cy = cb[NPTS + n], cz = cb[2 * NPTS + n];
    float dx = cb[j] - cx, dy = cb[NPTS + j] - cy, dz = cb[2 * NPTS + j] - cz;
    float rx = fb[n], ry = fb[NPTS + n], rz = fb[2 * NPTS + n];
    float ix = fb[j], iy = fb[NPTS + j], iz = fb[2 * NPTS + j];

    float nr_d  = angle3(rx, ry, rz, dx, dy, dz);
    float ni_d  = angle3(ix, iy, iz, dx, dy, dz);
    float nr_ni = angle3(rx, ry, rz, ix, iy, iz);
    float dn    = sqrtf(dx * dx + dy * dy + dz * dz);

    half_t* row = F + gid * 32;
    v8h lo, hi, z = {};
    lo[0] = (half_t)cx;    lo[1] = (half_t)cy;    lo[2] = (half_t)cz;
    lo[3] = (half_t)dx;    lo[4] = (half_t)dy;    lo[5] = (half_t)dz;
    lo[6] = (half_t)nr_d;  lo[7] = (half_t)ni_d;
    hi = z;
    hi[0] = (half_t)nr_ni; hi[1] = (half_t)dn;
    *(v8h*)(row)      = lo;
    *(v8h*)(row + 8)  = hi;
    *(v8h*)(row + 16) = z;
    *(v8h*)(row + 24) = z;
}

// ---------------- WMMA channel GEMM ----------------
// Xin: [BP][CIN] f16 ; W: [COUT][CIN] f16 ; Xout: [BP][COUT] f16
// Weights staged into LDS once per block (async copy when available).
// Octiles processed in groups of 4 independent accumulators so WMMAs issue
// back-to-back (no WMMA->VALU hazard NOPs between them).
#define PPW 4
template <int CIN, int COUT>
__global__ __launch_bounds__(256) void gemm_kernel(const half_t* __restrict__ Xin,
                                                   const half_t* __restrict__ W,
                                                   half_t* __restrict__ Xout) {
    constexpr int KS = CIN / 32;
    constexpr int OT = COUT / 16;
    extern __shared__ half_t wlds[];                 // COUT*CIN halves

    // ---- stage weights into LDS (16B per thread per iteration) ----
    for (int o = threadIdx.x * 8; o < COUT * CIN; o += 256 * 8) {
#ifdef HAVE_ASYNC_LDS
        __builtin_amdgcn_global_load_async_to_lds_b128(
            (gv4i_ptr)(const_cast<half_t*>(W) + o),
            (lv4i_ptr)(unsigned int)(unsigned long long)(wlds + o),
            0, 0);
#else
        *(v8h*)(wlds + o) = *(const v8h*)(W + o);
#endif
    }
#ifdef HAVE_ASYNC_LDS
#if __has_builtin(__builtin_amdgcn_s_wait_asynccnt)
    __builtin_amdgcn_s_wait_asynccnt(0);
#else
    asm volatile("s_wait_asynccnt 0" ::: "memory");
#endif
#endif
    __syncthreads();

    const int wave = threadIdx.x >> 5;
    const int lane = threadIdx.x & 31;
    const int li   = lane & 15;            // column within tile / A-row within tile
    const int g    = (lane >> 4) & 1;      // half-wave group
    const long long block_pt0 = (long long)blockIdx.x * (8 * PPW);

    for (int it = 0; it < PPW; ++it) {
        long long pt = block_pt0 + (long long)wave * PPW + it;
        long long p0 = pt * 16;
        const half_t* brow = Xin + (p0 + li) * CIN + g * 16;
        if (it + 1 < PPW) __builtin_prefetch(brow + 16 * CIN, 0, 1);  // global_prefetch_b8

        v16h bfrag[KS];
#pragma unroll
        for (int ks = 0; ks < KS; ++ks)
            bfrag[ks] = *(const v16h*)(brow + ks * 32);

#pragma unroll
        for (int og = 0; og < OT; og += 4) {
            v8f acc[4] = {};
#pragma unroll
            for (int ks = 0; ks < KS; ++ks) {
#pragma unroll
                for (int oi = 0; oi < 4; ++oi) {
                    const half_t* arow = wlds + ((og + oi) * 16 + li) * CIN + ks * 32 + g * 16;
                    v8h alo = *(const v8h*)(arow);           // ds_load_b128
                    v8h ahi = *(const v8h*)(arow + 8);
                    v16h a;
#pragma unroll
                    for (int q = 0; q < 8; ++q) { a[q] = alo[q]; a[q + 8] = ahi[q]; }
                    acc[oi] = __builtin_amdgcn_wmma_f32_16x16x32_f16(
                        false, a, false, bfrag[ks], (short)0, acc[oi], false, false);
                }
            }
#pragma unroll
            for (int oi = 0; oi < 4; ++oi) {
                v8h dout;
#pragma unroll
                for (int r = 0; r < 8; ++r) dout[r] = (half_t)acc[oi][r];
                *(v8h*)(Xout + (p0 + li) * COUT + (og + oi) * 16 + g * 8) = dout;
            }
        }
    }
}

// ---------------- column reduce: per-(batch,channel) sum & sum^2 ----------------
#define RPB 4096   // rows per block (divides PP)
template <int C>
__global__ __launch_bounds__(256) void reduce_kernel(const half_t* __restrict__ X,
                                                     float* __restrict__ gsum,
                                                     float* __restrict__ gsq) {
    __shared__ float s_sum[C];
    __shared__ float s_sq[C];
    for (int t = threadIdx.x; t < C; t += 256) { s_sum[t] = 0.f; s_sq[t] = 0.f; }
    __syncthreads();

    const long long row0 = (long long)blockIdx.x * RPB;
    const int b  = (int)(row0 / PP);
    const int c0 = (threadIdx.x * 8) % C;
    const int r0 = (threadIdx.x * 8) / C;
    const int rstep = 256 * 8 / C;

    float acc[8] = {0.f}, accq[8] = {0.f};
    for (int r = r0; r < RPB; r += rstep) {
        v8h v = *(const v8h*)(X + (row0 + r) * C + c0);
#pragma unroll
        for (int i = 0; i < 8; ++i) {
            float x = (float)v[i];
            acc[i] += x;
            accq[i] += x * x;
        }
    }
#pragma unroll
    for (int i = 0; i < 8; ++i) {
        atomicAdd(&s_sum[c0 + i], acc[i]);
        atomicAdd(&s_sq[c0 + i], accq[i]);
    }
    __syncthreads();
    for (int t = threadIdx.x; t < C; t += 256) {
        atomicAdd(&gsum[b * C + t], s_sum[t]);
        atomicAdd(&gsq[b * C + t], s_sq[t]);
    }
}

// ---------------- stats: mean + inv-std ----------------
__global__ void stats_kernel(const float* __restrict__ sum, const float* __restrict__ sq,
                             float* __restrict__ mean, float* __restrict__ istd, int n) {
    int t = blockIdx.x * 256 + threadIdx.x;
    if (t >= n) return;
    const float cnt = (float)PP;
    float m = sum[t] / cnt;
    float v = sq[t] / cnt - m * m;
    mean[t] = m;
    istd[t] = rsqrtf(v + EPSV);
}

// ---------------- normalize + relu (in place, x8 vectorized) ----------------
template <int C>
__global__ __launch_bounds__(256) void norm_relu_kernel(half_t* __restrict__ X,
                                                        const float* __restrict__ mean,
                                                        const float* __restrict__ istd) {
    long long t  = (long long)blockIdx.x * 256 + threadIdx.x;   // over BP*C/8
    long long e0 = t * 8;
    long long row = e0 / C;
    int c0 = (int)(e0 % C);
    int b  = (int)(row / PP);
    v8h v = *(v8h*)(X + e0);
#pragma unroll
    for (int i = 0; i < 8; ++i) {
        float x = (float)v[i];
        float y = (x - mean[b * C + c0 + i]) * istd[b * C + c0 + i];
        v[i] = (half_t)fmaxf(y, 0.f);
    }
    *(v8h*)(X + e0) = v;
}

// ---------------- final: norm + relu + max over K (vectorized x8) ----------------
__global__ __launch_bounds__(256) void final_kernel(const half_t* __restrict__ X2,
                                                    const float* __restrict__ mean,
                                                    const float* __restrict__ istd,
                                                    float* __restrict__ out) {
    const int sub  = threadIdx.x >> 5;               // 8 points per block
    const int lane = threadIdx.x & 31;               // 32 lanes x 8 channels = 256
    const int bn   = blockIdx.x * 8 + sub;
    const int n    = bn & (NPTS - 1);
    const int b    = bn >> 12;
    const int c0   = lane * 8;

    float m[8], s[8], best[8];
#pragma unroll
    for (int i = 0; i < 8; ++i) {
        m[i] = mean[b * 256 + c0 + i];
        s[i] = istd[b * 256 + c0 + i];
        best[i] = 0.f;                               // relu lower bound
    }
    const half_t* base = X2 + ((long long)bn * KNN) * 256 + c0;
    for (int k = 0; k < KNN; ++k) {
        v8h v = *(const v8h*)(base + (long long)k * 256);
#pragma unroll
        for (int i = 0; i < 8; ++i)
            best[i] = fmaxf(best[i], ((float)v[i] - m[i]) * s[i]);
    }
#pragma unroll
    for (int i = 0; i < 8; ++i)
        out[((long long)b * 256 + c0 + i) * NPTS + n] = best[i];
}

// ---------------- launch ----------------
extern "C" void kernel_launch(void* const* d_in, const int* in_sizes, int n_in,
                              void* d_out, int out_size, void* d_ws, size_t ws_size,
                              hipStream_t stream) {
    const float* coords = (const float*)d_in[0];
    const float* feats  = (const float*)d_in[1];
    const float* W0     = (const float*)d_in[2];
    const float* W1     = (const float*)d_in[3];
    const float* W2     = (const float*)d_in[4];
    float* out = (float*)d_out;

    char* w = (char*)d_ws;
    int*    idx   = (int*)(w + OFF_IDX);
    float*  stats = (float*)(w + OFF_STATS);
    float* sum0 = stats;            float* sq0 = sum0 + NB * 64;
    float* mean0 = sq0 + NB * 64;   float* istd0 = mean0 + NB * 64;
    float* sum1 = istd0 + NB * 64;  float* sq1 = sum1 + NB * 128;
    float* mean1 = sq1 + NB * 128;  float* istd1 = mean1 + NB * 128;
    float* sum2 = istd1 + NB * 128; float* sq2 = sum2 + NB * 256;
    float* mean2 = sq2 + NB * 256;  float* istd2 = mean2 + NB * 256;

    half_t* W0p = (half_t*)(w + OFF_W);
    half_t* W1p = W0p + 64 * 32;
    half_t* W2p = W1p + 128 * 64;

    half_t* bufA = (half_t*)(w + OFF_A);   // F0 (BPx32), then X1 (BPx128)
    half_t* bufB = (half_t*)(w + OFF_B);   // X0 (BPx64), then X2 (BPx256)

    const int nstats = NB * 4 * (64 + 128 + 256);
    zero_kernel<<<(nstats + 255) / 256, 256, 0, stream>>>(stats, nstats);
    wpad_kernel<<<(43008 + 255) / 256, 256, 0, stream>>>(W0, W1, W2, W0p, W1p, W2p);
    knn_kernel<<<NB * NPTS / 8, 256, 0, stream>>>(coords, idx);
    features_kernel<<<BP / 256, 256, 0, stream>>>(coords, feats, idx, bufA);

    const int gblocks = (BP / 16) / (8 * PPW);   // 2048
    const int rblocks = BP / RPB;                // 256

    // layer 0: 32(pad from 10) -> 64
    gemm_kernel<32, 64><<<gblocks, 256, 64 * 32 * 2, stream>>>(bufA, W0p, bufB);
    reduce_kernel<64><<<rblocks, 256, 0, stream>>>(bufB, sum0, sq0);
    stats_kernel<<<(NB * 64 + 255) / 256, 256, 0, stream>>>(sum0, sq0, mean0, istd0, NB * 64);
    norm_relu_kernel<64><<<(int)((long long)BP * 64 / 8 / 256), 256, 0, stream>>>(bufB, mean0, istd0);
    // layer 1: 64 -> 128
    gemm_kernel<64, 128><<<gblocks, 256, 128 * 64 * 2, stream>>>(bufB, W1p, bufA);
    reduce_kernel<128><<<rblocks, 256, 0, stream>>>(bufA, sum1, sq1);
    stats_kernel<<<(NB * 128 + 255) / 256, 256, 0, stream>>>(sum1, sq1, mean1, istd1, NB * 128);
    norm_relu_kernel<128><<<(int)((long long)BP * 128 / 8 / 256), 256, 0, stream>>>(bufA, mean1, istd1);
    // layer 2: 128 -> 256
    gemm_kernel<128, 256><<<gblocks, 256, 256 * 128 * 2, stream>>>(bufA, W2p, bufB);
    reduce_kernel<256><<<rblocks, 256, 0, stream>>>(bufB, sum2, sq2);
    stats_kernel<<<(NB * 256 + 255) / 256, 256, 0, stream>>>(sum2, sq2, mean2, istd2, NB * 256);
    // fused norm + relu + max over K
    final_kernel<<<NB * NPTS / 8, 256, 0, stream>>>(bufB, mean2, istd2, out);
}
